// AttentionPool_71459665871026
// MI455X (gfx1250) — matrix operand order repaired and checked by reading, not compiled, not run on hardware
//
#include <hip/hip_runtime.h>
#include <stdint.h>
#include <math.h>

typedef __attribute__((ext_vector_type(16))) _Float16 v16h;
typedef __attribute__((ext_vector_type(8)))  float    v8f;

#define D_DIM   128
#define NSPOT   4096
#define KWORDS  (NSPOT / 32)   // 128 bitmap words per gene row

#define GENES_PER_BLK 16
#define KCHUNK   512           // spots per LDS A-chunk
#define KC_WORDS (KCHUNK / 32) // 16 bitmap words per chunk per gene
#define A_STRIDE 520           // halves per gene row in LDS (8-half pad: bank spread)

// ---------------------------------------------------------------------------
// Kernel 1: logits[s] = v . tanh(W @ emb[s] + b)          (4096 x 128 threads)
// ---------------------------------------------------------------------------
__global__ void logits_kernel(const float* __restrict__ emb,
                              const float* __restrict__ W,
                              const float* __restrict__ bias,
                              const float* __restrict__ v,
                              float* __restrict__ logits)
{
    __shared__ float sh[D_DIM];
    __shared__ float red[D_DIM];
    const int s = blockIdx.x;
    const int t = threadIdx.x;            // 0..127
    sh[t] = emb[(size_t)s * D_DIM + t];
    __syncthreads();
    const float* wr = W + (size_t)t * D_DIM;
    float acc = bias[t];
#pragma unroll 8
    for (int k = 0; k < D_DIM; ++k) acc = fmaf(wr[k], sh[k], acc);
    red[t] = tanhf(acc) * v[t];
    __syncthreads();
    for (int off = D_DIM / 2; off > 0; off >>= 1) {
        if (t < off) red[t] += red[t + off];
        __syncthreads();
    }
    if (t == 0) logits[s] = red[0];
}

// ---------------------------------------------------------------------------
// Kernel 2: global max of logits (keeps exp() in (0,1] -> f16 safe)
// ---------------------------------------------------------------------------
__global__ void gmax_kernel(const float* __restrict__ logits,
                            float* __restrict__ gmax)
{
    __shared__ float red[256];
    float m = -INFINITY;
    for (int i = threadIdx.x; i < NSPOT; i += 256) m = fmaxf(m, logits[i]);
    red[threadIdx.x] = m;
    __syncthreads();
    for (int off = 128; off > 0; off >>= 1) {
        if (threadIdx.x < off)
            red[threadIdx.x] = fmaxf(red[threadIdx.x], red[threadIdx.x + off]);
        __syncthreads();
    }
    if (threadIdx.x == 0) gmax[0] = red[0];
}

// ---------------------------------------------------------------------------
// Kernel 3: E_f[s] = float(f16(exp(logit-gmax)));  Pt[d][s] = f16(E_f*emb[s][d])
// grid: (NSPOT/256, D_DIM) x 256  — Pt stored transposed [D][NSPOT]
// ---------------------------------------------------------------------------
__global__ void expand_kernel(const float* __restrict__ emb,
                              const float* __restrict__ logits,
                              const float* __restrict__ gmax,
                              float* __restrict__ E_f,
                              _Float16* __restrict__ Pt)
{
    const int s = blockIdx.x * 256 + threadIdx.x;
    const int d = blockIdx.y;
    const float    e  = expf(logits[s] - gmax[0]);
    const _Float16 eh = (_Float16)e;            // round once; den uses same value
    const float    ef = (float)eh;
    if (d == 0) E_f[s] = ef;
    Pt[(size_t)d * NSPOT + s] = (_Float16)(ef * emb[(size_t)s * D_DIM + d]);
}

// ---------------------------------------------------------------------------
// Kernel 4: zero the presence bitmap
// ---------------------------------------------------------------------------
__global__ void zero_kernel(uint32_t* __restrict__ p, int n)
{
    int i = blockIdx.x * 256 + threadIdx.x;
    if (i < n) p[i] = 0u;
}

// ---------------------------------------------------------------------------
// Kernel 5: scatter edges -> bitmap (atomicOr dedups duplicate (g,s) pairs,
// matching the reference's dense .set semantics)
// ---------------------------------------------------------------------------
__global__ void scatter_kernel(const int* __restrict__ gid,
                               const int* __restrict__ sid,
                               uint32_t* __restrict__ bitmap, int n_edges)
{
    int i = blockIdx.x * 256 + threadIdx.x;
    if (i < n_edges) {
        const int g = gid[i], s = sid[i];
        atomicOr(&bitmap[(size_t)g * KWORDS + (s >> 5)], 1u << (s & 31));
    }
}

// ---------------------------------------------------------------------------
// Kernel 6: inv_den[g] = 1 / sum_{s in S_g} E_f[s]   (0 for empty rows)
// ---------------------------------------------------------------------------
__global__ void den_kernel(const uint32_t* __restrict__ bitmap,
                           const float* __restrict__ E_f,
                           float* __restrict__ inv_den)
{
    __shared__ float red[KWORDS];
    const int g = blockIdx.x, t = threadIdx.x;
    uint32_t w = bitmap[(size_t)g * KWORDS + t];
    const float* ev = E_f + t * 32;
    float acc = 0.f;
    while (w) {
        int b = __ffs(w) - 1;
        acc += ev[b];
        w &= (w - 1u);
    }
    red[t] = acc;
    __syncthreads();
    for (int off = KWORDS / 2; off > 0; off >>= 1) {
        if (t < off) red[t] += red[t + off];
        __syncthreads();
    }
    if (t == 0) inv_den[g] = red[0] > 0.f ? 1.f / red[0] : 0.f;
}

// ---------------------------------------------------------------------------
// Kernel 7: out = diag(inv_den) * (M @ Pt^T) via v_wmma_f32_16x16x32_f16.
//
// Block = 256 threads = 8 waves covering 16 genes x 128 dims; wave w owns
// N-tile [16w,16w+16). The binary A tile (16 genes x 512 K) is expanded from
// the bitmap into LDS ONCE per block per chunk (amortized 8x across waves),
// then each wave reads its A fragment with two ds_load_b128 per WMMA step:
//   A layout (16-bit 16x32): lanes 0-15 (M=l) need K 0-7 & 16-23,
//   lanes 16-31 need K 8-15 & 24-31  -> two contiguous 16B runs per lane.
// B (32x16 f16) comes from transposed Pt[d][s]: lane = column N; lanes 0-15
// hold K 0-7 & 16-23, lanes 16-31 hold K 8-15 & 24-31 -> two 16B runs per lane
// (two global_load_b128, L2-resident).
// ---------------------------------------------------------------------------
__global__ void __launch_bounds__(256)
attn_gemm_kernel(const uint32_t* __restrict__ bitmap,
                 const _Float16* __restrict__ Pt,      // [D_DIM][NSPOT]
                 const float* __restrict__ inv_den,
                 float* __restrict__ out)
{
    __shared__ uint32_t bm[GENES_PER_BLK * KWORDS];     // 8 KB
    __shared__ _Float16 As[GENES_PER_BLK * A_STRIDE];   // 16.25 KB

    const int tid  = threadIdx.x;
    const int lane = tid & 31;
    const int wave = tid >> 5;                 // 0..7 -> N tile
    const int l    = lane & 15;
    const int half = lane >> 4;
    const int g_base = blockIdx.x << 4;        // 16 genes per block
    const int dcol   = (wave << 4) + l;        // output dim column

    // coalesced preload of this block's 16 bitmap rows (2048 words)
    const uint32_t* bsrc = bitmap + (size_t)g_base * KWORDS;
#pragma unroll
    for (int i = 0; i < 8; ++i)
        bm[tid + 256 * i] = bsrc[tid + 256 * i];

    // expansion assignment: thread -> (gene eg, chunk-word ew)
    const int eg = tid & 15;
    const int ew = tid >> 4;                   // 0..15

    const _Float16* prow = Pt + (size_t)dcol * NSPOT + (half << 3);

    v8f c = {};
    for (int ch = 0; ch < NSPOT / KCHUNK; ++ch) {
        __syncthreads();   // prior consume done (and bm preload on ch==0)

        // expand one bitmap word -> 32 f16 (16 packed pairs, 4x b128 stores)
        const uint32_t w = bm[eg * KWORDS + ch * KC_WORDS + ew];
        _Float16* dst = As + eg * A_STRIDE + ew * 32;
#pragma unroll
        for (int q = 0; q < 4; ++q) {
            uint4 val;
            uint32_t* vp = (uint32_t*)&val;
#pragma unroll
            for (int r = 0; r < 4; ++r) {
                const uint32_t t = (w >> ((q * 4 + r) * 2)) & 3u;
                vp[r] = ((t & 1u) * 0x3C00u) | ((t >> 1) * 0x3C000000u);
            }
            *(uint4*)(dst + q * 8) = val;
        }
        __syncthreads();

        const _Float16* arow  = As + l * A_STRIDE + (half << 3);
        const _Float16* bsrcp = prow + ch * KCHUNK;
#pragma unroll 4
        for (int sk = 0; sk < KC_WORDS; ++sk) {
            union { v16h v; uint4 q[2]; } A;
            A.q[0] = *(const uint4*)(arow + sk * 32);        // K 0-7  (+8 if half)
            A.q[1] = *(const uint4*)(arow + sk * 32 + 16);   // K 16-23(+8 if half)
            union { v16h v; uint4 q[2]; } B;
            const uint4* bp = (const uint4*)(bsrcp + sk * 32);
            B.q[0] = bp[0];
            B.q[1] = bp[2];
            c = __builtin_amdgcn_wmma_f32_16x16x32_f16(false, A.v, false, B.v,
                                                       (short)0, c, false, false);
        }
    }

    const int gm = g_base + (half << 3);
#pragma unroll
    for (int i = 0; i < 8; ++i) {
        const int g = gm + i;                  // row M = i + 8*half
        out[(size_t)g * D_DIM + dcol] = c[i] * inv_den[g];
    }
}

// ---------------------------------------------------------------------------
extern "C" void kernel_launch(void* const* d_in, const int* in_sizes, int n_in,
                              void* d_out, int out_size, void* d_ws, size_t ws_size,
                              hipStream_t stream)
{
    const float* emb = (const float*)d_in[0];   // [4096,128]
    const float* W   = (const float*)d_in[1];   // [128,128]
    const float* b   = (const float*)d_in[2];   // [128]
    const float* v   = (const float*)d_in[3];   // [128]
    const int*   gid = (const int*)d_in[4];     // [n_edges]
    const int*   sid = (const int*)d_in[5];     // [n_edges]
    const int n_edges = in_sizes[4];
    const int n_genes = out_size / D_DIM;       // 20000
    float* out = (float*)d_out;

    // carve 256B-aligned workspace (~11.4 MB total)
    uintptr_t p = (uintptr_t)d_ws;
    auto carve = [&](size_t bytes) -> uintptr_t {
        uintptr_t q = (p + 255) & ~(uintptr_t)255;
        p = q + bytes;
        return q;
    };
    float*    logits = (float*)   carve((size_t)NSPOT * 4);
    float*    gmax   = (float*)   carve(4);
    float*    E_f    = (float*)   carve((size_t)NSPOT * 4);
    float*    invden = (float*)   carve((size_t)n_genes * 4);
    uint32_t* bitmap = (uint32_t*)carve((size_t)n_genes * KWORDS * 4);
    _Float16* Pt     = (_Float16*)carve((size_t)D_DIM * NSPOT * 2);
    (void)ws_size; (void)n_in;

    logits_kernel<<<NSPOT, D_DIM, 0, stream>>>(emb, W, b, v, logits);
    gmax_kernel<<<1, 256, 0, stream>>>(logits, gmax);
    expand_kernel<<<dim3(NSPOT / 256, D_DIM), 256, 0, stream>>>(emb, logits, gmax, E_f, Pt);

    const int nwords = n_genes * KWORDS;
    zero_kernel<<<(nwords + 255) / 256, 256, 0, stream>>>(bitmap, nwords);
    scatter_kernel<<<(n_edges + 255) / 256, 256, 0, stream>>>(gid, sid, bitmap, n_edges);
    den_kernel<<<n_genes, KWORDS, 0, stream>>>(bitmap, E_f, invden);

    attn_gemm_kernel<<<n_genes / 16, 256, 0, stream>>>(bitmap, Pt, invden, out);
}
